// FreqFusion_28862180229162
// MI455X (gfx1250) — matrix operand order repaired and checked by reading, not compiled, not run on hardware
//
#include <hip/hip_runtime.h>
#include <hip/hip_bf16.h>
#include <stdint.h>

// ---------------------------------------------------------------------------
// FreqFusion on MI455X (gfx1250, wave32). Fixed geometry:
//   B=4, Chr=Clr=256, H=W=128, comp=64, K=3.
//   k1: conv1x1 256->64 as BF16 WMMA GEMM   (hr_feat -> compressed)
//   k2: conv3x3 64->9 + softmax*hamming norm (compressed -> mask)
//   k3: CARAFE(3x3, reflect) + residual + 2x2 avg pool (hr_feat,mask -> down)
//   k4: conv1x1 256->256 as BF16 WMMA GEMM + lr_feat   (down -> out)
// Memory-bound (~6 GFLOP vs ~150 MB traffic, all L2-resident on 192MB L2).
// GEMMs: one wave = 16(M) x 32(N) tile, K=256 fully unrolled (8 x K32 steps,
// 2 WMMAs per step sharing one A fragment). All strides compile-time so the
// B-tile loads fold into immediate-offset load clauses (no u64 add chains).
// ---------------------------------------------------------------------------

typedef __attribute__((ext_vector_type(16))) __bf16 v16bf;
typedef __attribute__((ext_vector_type(8)))  float  v8f;

template <int CIN, int COUT, int NPIX, bool HAS_ADD>
__global__ void __launch_bounds__(256)
conv1x1_wmma(const float* __restrict__ X,      // [B, CIN, NPIX]
             const float* __restrict__ Wt,     // [COUT, CIN]
             const float* __restrict__ bias,   // [COUT]
             const float* __restrict__ addend, // [B, COUT, NPIX] (if HAS_ADD)
             float* __restrict__ Y) {          // [B, COUT, NPIX]
  constexpr int N2T = NPIX / 32;               // 32-wide N blocks (2 WMMA tiles)
  constexpr int MT  = COUT / 16;
  const int wave = (blockIdx.x * blockDim.x + threadIdx.x) >> 5;
  const int lane = threadIdx.x & 31;
  const int nt2 = wave % N2T;
  const int mt  = (wave / N2T) % MT;
  const int bb  = wave / (N2T * MT);           // grid is exact; wave-uniform

  const int m  = lane & 15;                    // M row (A) / N col (B,C)
  const int hi = lane >> 4;                    // lane half
  const int n0 = nt2 * 32;

  // Per-lane bases; all further offsets are compile-time immediates.
  const float* xk   = X + (size_t)bb * CIN * NPIX + (size_t)hi * 8 * NPIX + n0 + m;
  const float* wrow = Wt + (mt * 16 + m) * CIN + hi * 8;

  v8f acc0 = {0.f, 0.f, 0.f, 0.f, 0.f, 0.f, 0.f, 0.f};
  v8f acc1 = {0.f, 0.f, 0.f, 0.f, 0.f, 0.f, 0.f, 0.f};

#pragma unroll
  for (int kk = 0; kk < CIN; kk += 32) {
    if (kk + 32 < CIN)  // next K-slab toward L2 (global_prefetch_b8)
      __builtin_prefetch(xk + (size_t)(kk + 32) * NPIX, 0, 1);

    // A fragment: 16x32 bf16 weights (ISA 16-bit A layout):
    // lane half hi contributes K base 8*hi; element e -> K (e<8 ? e : e+8).
    // Per lane the two 8-float runs are contiguous -> b128 load pairs.
    v16bf a, b0, b1;
#pragma unroll
    for (int e = 0; e < 16; ++e)
      a[e] = (__bf16)wrow[kk + (e < 8 ? e : e + 8)];

    // B fragments: two 32x16 bf16 activation tiles (same K order as A:
    // element e -> k = 8*hi + (e<8 ? e : e+8)). Row offsets
    // are immediates; per element each lane-half reads one
    // contiguous 64B row segment.
#pragma unroll
    for (int e = 0; e < 16; ++e) {
      const float* xr = xk + (size_t)(kk + (e < 8 ? e : e + 8)) * NPIX;
      b0[e] = (__bf16)xr[0];
      b1[e] = (__bf16)xr[16];
    }
    acc0 = __builtin_amdgcn_wmma_f32_16x16x32_bf16(false, a, false, b0,
                                                   (short)0, acc0, false, false);
    acc1 = __builtin_amdgcn_wmma_f32_16x16x32_bf16(false, a, false, b1,
                                                   (short)0, acc1, false, false);
  }

  // C/D layout: VGPR r -> (M = r + 8*hi, N = lane&15)
  const int mo0 = mt * 16 + hi * 8;
  const float* bp = bias + mo0;
  float* yp = Y + ((size_t)bb * COUT + mo0) * NPIX + n0 + m;
  const float* ap = HAS_ADD ? addend + ((size_t)bb * COUT + mo0) * NPIX + n0 + m
                            : nullptr;
#pragma unroll
  for (int r = 0; r < 8; ++r) {
    const float bv = bp[r];
    float v0 = acc0[r] + bv;
    float v1 = acc1[r] + bv;
    if (HAS_ADD) {
      v0 += ap[(size_t)r * NPIX];
      v1 += ap[(size_t)r * NPIX + 16];
    }
    yp[(size_t)r * NPIX]      = v0;
    yp[(size_t)r * NPIX + 16] = v1;
  }
}

// conv3x3 (64->9, zero pad) + softmax over 9 taps + hamming reweight + renorm.
// Geometry hardcoded: B=4, Cin=64, H=W=128.
__global__ void __launch_bounds__(256)
mask_kernel(const float* __restrict__ comp, // [4,64,128,128]
            const float* __restrict__ Wc,   // [9,64,3,3]
            const float* __restrict__ bc,   // [9]
            float* __restrict__ mask) {     // [4,9,128,128]
  constexpr int H = 128, W = 128, HW = H * W;
  __shared__ float sW[9 * 64 * 9];          // 20.25 KB of the 320 KB WGP LDS
  for (int i = threadIdx.x; i < 9 * 64 * 9; i += blockDim.x) sW[i] = Wc[i];
  __syncthreads();

  const int gid = blockIdx.x * blockDim.x + threadIdx.x;  // 4*16384 threads
  const int bb = gid >> 14;
  const int p  = gid & (HW - 1);
  const int y  = p >> 7, x = p & (W - 1);

  float acc[9];
#pragma unroll
  for (int o = 0; o < 9; ++o) acc[o] = bc[o];

  const float* cb = comp + (size_t)bb * 64 * HW;
  for (int c = 0; c < 64; ++c) {
    float in[9];
#pragma unroll
    for (int t = 0; t < 9; ++t) {
      const int ty = y + (t / 3) - 1, tx = x + (t % 3) - 1;
      in[t] = (ty >= 0 && ty < H && tx >= 0 && tx < W)
                  ? cb[(size_t)c * HW + ty * W + tx] : 0.f;
    }
#pragma unroll
    for (int o = 0; o < 9; ++o) {
      const float* wp = &sW[(o * 64 + c) * 9];
#pragma unroll
      for (int t = 0; t < 9; ++t) acc[o] = fmaf(in[t], wp[t], acc[o]);
    }
  }

  // softmax over the 9 taps, hamming(3)^T hamming(3) reweight, renormalize
  const float ham[9] = {0.0064f, 0.08f, 0.0064f,
                        0.08f,   1.0f,  0.08f,
                        0.0064f, 0.08f, 0.0064f};
  float mx = acc[0];
#pragma unroll
  for (int o = 1; o < 9; ++o) mx = fmaxf(mx, acc[o]);
  float s = 0.f;
#pragma unroll
  for (int o = 0; o < 9; ++o) { acc[o] = __expf(acc[o] - mx); s += acc[o]; }
  const float inv = 1.f / s;
  float s2 = 0.f;
#pragma unroll
  for (int o = 0; o < 9; ++o) { acc[o] *= inv * ham[o]; s2 += acc[o]; }
  const float inv2 = 1.f / s2;
#pragma unroll
  for (int o = 0; o < 9; ++o)
    mask[((size_t)bb * 9 + o) * HW + p] = acc[o] * inv2;
}

// Fused: hr_hf = hr - carafe(hr, mask); 2x2 average pool -> down. Reflect pad.
// Geometry hardcoded: B=4, C=256, H=W=128 -> down [4,256,64,64].
__global__ void __launch_bounds__(256)
carafe_pool_kernel(const float* __restrict__ hr,   // [4,256,128,128]
                   const float* __restrict__ mask, // [4,9,128,128]
                   float* __restrict__ down) {     // [4,256,64,64]
  constexpr int H = 128, W = 128, HW = H * W;
  constexpr int Wd = 64, HWd = 64 * 64;
  const int gid = blockIdx.x * blockDim.x + threadIdx.x;  // 2^22 threads
  const int pd = gid & (HWd - 1);
  const int c  = (gid >> 12) & 255;
  const int bb = gid >> 20;
  const int yd = pd >> 6, xd = pd & (Wd - 1);

  const float* hrc = hr + ((size_t)bb * 256 + c) * HW;
  const float* mb  = mask + (size_t)bb * 9 * HW;   // 2.4 MB total: L2-resident

  float sum = 0.f;
#pragma unroll
  for (int dy = 0; dy < 2; ++dy) {
#pragma unroll
    for (int dx = 0; dx < 2; ++dx) {
      const int y = 2 * yd + dy, x = 2 * xd + dx;
      float v = 0.f;
#pragma unroll
      for (int t = 0; t < 9; ++t) {
        int ys = y + t / 3 - 1; ys = ys < 0 ? -ys : (ys >= H ? 2 * H - 2 - ys : ys);
        int xs = x + t % 3 - 1; xs = xs < 0 ? -xs : (xs >= W ? 2 * W - 2 - xs : xs);
        v = fmaf(hrc[ys * W + xs], mb[(size_t)t * HW + y * W + x], v);
      }
      sum += hrc[y * W + x] - v;
    }
  }
  down[gid] = sum * 0.25f;  // [4,256,64,64] flattened == gid
}

extern "C" void kernel_launch(void* const* d_in, const int* in_sizes, int n_in,
                              void* d_out, int out_size, void* d_ws, size_t ws_size,
                              hipStream_t stream) {
  const float* hr   = (const float*)d_in[0];  // [4,256,128,128]
  const float* lr   = (const float*)d_in[1];  // [4,256,64,64]
  const float* Whrc = (const float*)d_in[2];  // [64,256]
  const float* bhrc = (const float*)d_in[3];  // [64]
  const float* Wenc = (const float*)d_in[4];  // [9,64,3,3]
  const float* benc = (const float*)d_in[5];  // [9]
  const float* Wad  = (const float*)d_in[6];  // [256,256]
  const float* bad  = (const float*)d_in[7];  // [256]
  float* out = (float*)d_out;                 // [4,256,64,64]

  constexpr int B = 4, CHR = 256, COMP = 64;
  constexpr int HW  = 128 * 128;  // 16384
  constexpr int HWd = 64 * 64;    // 4096

  // workspace: compressed (16.8MB) | mask (2.4MB) | down (16.8MB) ~= 36MB
  float* compressed = (float*)d_ws;
  float* mask = compressed + (size_t)B * COMP * HW;
  float* down = mask + (size_t)B * 9 * HW;

  // k1: compress 256->64 : waves = 4 * (64/16) * (16384/32) = 8192, 8 waves/blk
  conv1x1_wmma<CHR, COMP, HW, false><<<B * (COMP / 16) * (HW / 32) / 8, 256, 0,
                                       stream>>>(hr, Whrc, bhrc, nullptr,
                                                 compressed);
  // k2: conv3x3 + kernel normalizer (65536 threads)
  mask_kernel<<<B * HW / 256, 256, 0, stream>>>(compressed, Wenc, benc, mask);

  // k3: carafe + residual + 2x2 pool (2^22 threads)
  carafe_pool_kernel<<<B * CHR * HWd / 256, 256, 0, stream>>>(hr, mask, down);

  // k4: adapt 256->256 + lr : waves = 4 * (256/16) * (4096/32) = 8192
  conv1x1_wmma<CHR, CHR, HWd, true><<<B * (CHR / 16) * (HWd / 32) / 8, 256, 0,
                                      stream>>>(down, Wad, bad, lr, out);
}
